// HEATConv_64707977282141
// MI455X (gfx1250) — matrix operand order repaired and checked by reading, not compiled, not run on hardware
//
#include <hip/hip_runtime.h>
#include <math.h>

typedef __attribute__((ext_vector_type(16))) _Float16 v16h;
typedef __attribute__((ext_vector_type(8)))  float    v8f;

#define N_NODES 50000
#define E_EDGES 800000
#define CIN     64
#define HID     64
#define NHEAD   2
#define OC      32
#define EC      16

// ---------------- utility ----------------
__global__ void fill_kernel(float* __restrict__ p, float v, int n) {
  int i = blockIdx.x * blockDim.x + threadIdx.x;
  if (i < n) p[i] = v;
}

__device__ inline float atomicMaxF(float* addr, float val) {
  unsigned int* ua = (unsigned int*)addr;
  unsigned int old = *ua;
  while (true) {
    float cur = __uint_as_float(old);
    if (cur >= val) break;
    unsigned int assumed = old;
    old = atomicCAS(ua, assumed, __float_as_uint(val));
    if (old == assumed) break;
  }
  return __uint_as_float(old);
}

// ---------------- Fused WMMA QKV projection: {Q,K,V}[N,64] = X[N,64] @ {Wq,Wk,Wv}[64,64] ----------
// One wave (32 lanes, EXEC all-1s as WMMA requires) per 16-row tile. A fragments (16x32 f16,
// contiguous K map k=16*hi+e) are loaded ONCE and reused across the three weight matrices:
// 3 GEMMs x 4 N-tiles x 2 K-chunks = 24 v_wmma per wave.
__global__ void proj_qkv_kernel(const float* __restrict__ X,
                                const float* __restrict__ Wq,
                                const float* __restrict__ Wk,
                                const float* __restrict__ Wv,
                                float* __restrict__ Q,
                                float* __restrict__ K,
                                float* __restrict__ V) {
  const int lane = threadIdx.x;
  const int hi   = lane >> 4;
  const int ml   = lane & 15;
  const int tile = blockIdx.x;
  const int row  = tile * 16 + ml;

  v16h a0, a1;
#pragma unroll
  for (int e = 0; e < 16; ++e) {
    int v = e >> 1, hh = e & 1;
    int k = hi * 16 + e;   // A: linear K per lane half (matches B)
    a0[e] = (_Float16)X[(size_t)row * CIN + k];
    a1[e] = (_Float16)X[(size_t)row * CIN + 32 + k];
  }

  const float* Ws[3] = {Wq, Wk, Wv};
  float*       Ys[3] = {Q, K, V};

#pragma unroll
  for (int w = 0; w < 3; ++w) {
    const float* __restrict__ W = Ws[w];
    float*       __restrict__ Y = Ys[w];
#pragma unroll
    for (int t = 0; t < 4; ++t) {
      const int n = t * 16 + ml;
      v16h b0, b1;
#pragma unroll
      for (int e = 0; e < 16; ++e) {
        b0[e] = (_Float16)W[(hi * 16 + e) * HID + n];          // B: linear K per lane half
        b1[e] = (_Float16)W[(32 + hi * 16 + e) * HID + n];
      }
      v8f c = {};
      c = __builtin_amdgcn_wmma_f32_16x16x32_f16(false, a0, false, b0, (short)0, c, false, false);
      c = __builtin_amdgcn_wmma_f32_16x16x32_f16(false, a1, false, b1, (short)0, c, false, false);
#pragma unroll
      for (int r = 0; r < 8; ++r)
        Y[(size_t)(tile * 16 + hi * 8 + r) * HID + n] = c[r];  // D: M=hi*8+r, N=lane&15
    }
  }
}

// ---------------- Edge pass 1: attention scores + segment max ----------------
__global__ void edge_score_kernel(const float* __restrict__ qdst,
                                  const float* __restrict__ ksrc,
                                  const float* __restrict__ embs,
                                  const float* __restrict__ embd,
                                  const float* __restrict__ rel,
                                  const float* __restrict__ We,
                                  const float* __restrict__ eattr,
                                  const float* __restrict__ a_attn,
                                  const int*   __restrict__ row,
                                  const int*   __restrict__ col,
                                  float* __restrict__ sc,
                                  float* __restrict__ m) {
  __shared__ float sWe[EC * HID];
  __shared__ float sA[4 * OC];
  __shared__ float sRel[HID], sEs[HID], sEd[HID];
  for (int i = threadIdx.x; i < EC * HID; i += blockDim.x) sWe[i] = We[i];
  for (int i = threadIdx.x; i < 4 * OC;  i += blockDim.x) sA[i]  = a_attn[i];
  for (int i = threadIdx.x; i < HID;     i += blockDim.x) {
    sRel[i] = rel[i]; sEs[i] = embs[i]; sEd[i] = embd[i];
  }
  __syncthreads();

  int idx = blockIdx.x * blockDim.x + threadIdx.x;
  if (idx >= E_EDGES * NHEAD) return;
  const int e = idx >> 1, h = idx & 1;
  const int r = row[e], c = col[e];

  float ea[EC];
#pragma unroll
  for (int j = 0; j < EC; ++j) ea[j] = eattr[(size_t)e * EC + j];

  const float* q = qdst + (size_t)c * HID + h * OC;
  const float* k = ksrc + (size_t)r * HID + h * OC;
  float score = 0.f;
#pragma unroll 4
  for (int j = 0; j < OC; ++j) {
    float eb = 0.f;
#pragma unroll
    for (int cc = 0; cc < EC; ++cc) eb += ea[cc] * sWe[cc * HID + h * OC + j];
    float qd = q[j] + sEd[h * OC + j];
    float kk = k[j] + sEs[h * OC + j];
    float rb = sRel[h * OC + j];
    score += tanhf(qd) * sA[j] + tanhf(kk) * sA[OC + j]
           + tanhf(rb) * sA[2 * OC + j] + tanhf(eb) * sA[3 * OC + j];
  }
  sc[idx] = score;
  atomicMaxF(&m[c * NHEAD + h], score);
}

// ---------------- Edge pass 2: exp, segment sum, weighted message scatter ----------------
// Normalization folded to node level: agg accumulates ex*msg, s accumulates ex;
// the output kernel divides once per node (saves a full edge pass).
__global__ void edge_agg_kernel(const float* __restrict__ sc,
                                const float* __restrict__ m,
                                const float* __restrict__ vsrc,
                                const float* __restrict__ rel,
                                const float* __restrict__ We,
                                const float* __restrict__ eattr,
                                const int*   __restrict__ row,
                                const int*   __restrict__ col,
                                float* __restrict__ ssum,
                                float* __restrict__ agg) {
  __shared__ float sWe[EC * HID];
  __shared__ float sRel[HID];
  for (int i = threadIdx.x; i < EC * HID; i += blockDim.x) sWe[i] = We[i];
  for (int i = threadIdx.x; i < HID;      i += blockDim.x) sRel[i] = rel[i];
  __syncthreads();

  int idx = blockIdx.x * blockDim.x + threadIdx.x;
  if (idx >= E_EDGES * NHEAD) return;
  const int e = idx >> 1, h = idx & 1;
  const int r = row[e], c = col[e];

  float ea[EC];
#pragma unroll
  for (int j = 0; j < EC; ++j) ea[j] = eattr[(size_t)e * EC + j];

  const float ex = expf(sc[idx] - m[c * NHEAD + h]);
  atomicAdd(&ssum[c * NHEAD + h], ex);

  const float* v = vsrc + (size_t)r * HID + h * OC;
#pragma unroll 4
  for (int j = 0; j < OC; ++j) {
    float eb = 0.f;
#pragma unroll
    for (int cc = 0; cc < EC; ++cc) eb += ea[cc] * sWe[cc * HID + h * OC + j];
    float msg = v[j] + sRel[h * OC + j] + eb;
    atomicAdd(&agg[(size_t)c * HID + h * OC + j], ex * msg);
  }
}

// ---------------- Output WMMA GEMM: out = (agg/s) @ Wo + x @ Wr + bo ----------------
// One wave per 16-row tile; two 16x16 output tiles (OC=32); 8 WMMAs (2 K-chunks x 2 GEMMs x 2 N-tiles).
__global__ void out_gemm_kernel(const float* __restrict__ agg,
                                const float* __restrict__ ssum,
                                const float* __restrict__ X,
                                const float* __restrict__ Wo,
                                const float* __restrict__ bo,
                                const float* __restrict__ Wr,
                                float* __restrict__ out) {
  const int lane = threadIdx.x;
  const int hi   = lane >> 4;
  const int ml   = lane & 15;
  const int tile = blockIdx.x;
  const int row  = tile * 16 + ml;

  float s0 = ssum[row * NHEAD + 0];
  float s1 = ssum[row * NHEAD + 1];
  float inv0 = (s0 > 0.f) ? 1.f / s0 : 0.f;   // isolated node -> agg term 0 -> output x@Wr + bias
  float inv1 = (s1 > 0.f) ? 1.f / s1 : 0.f;

  v16h a0, a1, x0, x1;
#pragma unroll
  for (int e = 0; e < 16; ++e) {
    int v = e >> 1, hh = e & 1;
    int k = hi * 16 + e;
    a0[e] = (_Float16)(agg[(size_t)row * HID + k]      * inv0);  // head 0 block (K 0..31)
    a1[e] = (_Float16)(agg[(size_t)row * HID + 32 + k] * inv1);  // head 1 block (K 32..63)
    x0[e] = (_Float16)X[(size_t)row * CIN + k];
    x1[e] = (_Float16)X[(size_t)row * CIN + 32 + k];
  }

#pragma unroll
  for (int t = 0; t < 2; ++t) {
    const int n = t * 16 + ml;
    v16h bo0, bo1, br0, br1;
#pragma unroll
    for (int e = 0; e < 16; ++e) {
      bo0[e] = (_Float16)Wo[(hi * 16 + e) * OC + n];
      bo1[e] = (_Float16)Wo[(32 + hi * 16 + e) * OC + n];
      br0[e] = (_Float16)Wr[(hi * 16 + e) * OC + n];
      br1[e] = (_Float16)Wr[(32 + hi * 16 + e) * OC + n];
    }
    v8f c = {};
    c = __builtin_amdgcn_wmma_f32_16x16x32_f16(false, a0, false, bo0, (short)0, c, false, false);
    c = __builtin_amdgcn_wmma_f32_16x16x32_f16(false, a1, false, bo1, (short)0, c, false, false);
    c = __builtin_amdgcn_wmma_f32_16x16x32_f16(false, x0, false, br0, (short)0, c, false, false);
    c = __builtin_amdgcn_wmma_f32_16x16x32_f16(false, x1, false, br1, (short)0, c, false, false);
    const float bias = bo[n];
#pragma unroll
    for (int r = 0; r < 8; ++r)
      out[(size_t)(tile * 16 + hi * 8 + r) * OC + n] = c[r] + bias;
  }
}

// ---------------- launch ----------------
extern "C" void kernel_launch(void* const* d_in, const int* in_sizes, int n_in,
                              void* d_out, int out_size, void* d_ws, size_t ws_size,
                              hipStream_t stream) {
  (void)in_sizes; (void)n_in; (void)out_size; (void)ws_size;

  const float* x_a    = (const float*)d_in[0];
  const float* x_b    = (const float*)d_in[1];
  const float* eat1   = (const float*)d_in[2];
  const float* eat2   = (const float*)d_in[3];
  const float* Wq_a   = (const float*)d_in[4];
  const float* Wk_a   = (const float*)d_in[5];
  const float* Wv_a   = (const float*)d_in[6];
  const float* Wq_b   = (const float*)d_in[7];
  const float* Wk_b   = (const float*)d_in[8];
  const float* Wv_b   = (const float*)d_in[9];
  const float* emb_a  = (const float*)d_in[10];
  const float* emb_b  = (const float*)d_in[11];
  const float* rel1   = (const float*)d_in[12];
  const float* rel2   = (const float*)d_in[13];
  const float* We     = (const float*)d_in[14];
  const float* a_attn = (const float*)d_in[15];
  const float* Wo_a   = (const float*)d_in[16];
  const float* bo_a   = (const float*)d_in[17];
  const float* Wo_b   = (const float*)d_in[18];
  const float* bo_b   = (const float*)d_in[19];
  const float* Wr_a   = (const float*)d_in[20];
  const float* Wr_b   = (const float*)d_in[21];
  const int*   row1   = (const int*)d_in[22];
  const int*   col1   = (const int*)d_in[23];
  const int*   row2   = (const int*)d_in[24];
  const int*   col2   = (const int*)d_in[25];
  float* out = (float*)d_out;

  // workspace carve-up (floats): 6*NH + 2*EH + 4*NHh + 2*NH ~= 29.2M floats ~= 117 MB
  const size_t NH  = (size_t)N_NODES * HID;    // 3.2M
  const size_t EH  = (size_t)E_EDGES * NHEAD;  // 1.6M
  const size_t NHh = (size_t)N_NODES * NHEAD;  // 100k
  float* ws   = (float*)d_ws;
  float* q_a  = ws;            float* k_a = q_a + NH;  float* v_a = k_a + NH;
  float* q_b  = v_a + NH;      float* k_b = q_b + NH;  float* v_b = k_b + NH;
  float* sc1  = v_b + NH;      float* sc2 = sc1 + EH;
  float* m_b  = sc2 + EH;      float* s_b = m_b + NHh;
  float* m_a  = s_b + NHh;     float* s_a = m_a + NHh;
  float* agg_b = s_a + NHh;    float* agg_a = agg_b + NH;

  const int TILES = N_NODES / 16;           // 3125 exact
  const int EBLK  = 256;
  const int EGRID = (E_EDGES * NHEAD + EBLK - 1) / EBLK;
  const int FBLK  = 256;

  // init segment stats / accumulators
  fill_kernel<<<((int)NHh + FBLK - 1) / FBLK, FBLK, 0, stream>>>(m_a, -3.0e38f, (int)NHh);
  fill_kernel<<<((int)NHh + FBLK - 1) / FBLK, FBLK, 0, stream>>>(m_b, -3.0e38f, (int)NHh);
  fill_kernel<<<((int)NHh + FBLK - 1) / FBLK, FBLK, 0, stream>>>(s_a, 0.f, (int)NHh);
  fill_kernel<<<((int)NHh + FBLK - 1) / FBLK, FBLK, 0, stream>>>(s_b, 0.f, (int)NHh);
  fill_kernel<<<((int)NH + FBLK - 1) / FBLK, FBLK, 0, stream>>>(agg_a, 0.f, (int)NH);
  fill_kernel<<<((int)NH + FBLK - 1) / FBLK, FBLK, 0, stream>>>(agg_b, 0.f, (int)NH);

  // fused node projections (WMMA): X read once per tile, 24 v_wmma per wave
  proj_qkv_kernel<<<TILES, 32, 0, stream>>>(x_a, Wq_a, Wk_a, Wv_a, q_a, k_a, v_a);
  proj_qkv_kernel<<<TILES, 32, 0, stream>>>(x_b, Wq_b, Wk_b, Wv_b, q_b, k_b, v_b);

  // relation 1: a -> b (dst = b) ; relation 2: b -> a (dst = a)
  edge_score_kernel<<<EGRID, EBLK, 0, stream>>>(q_b, k_a, emb_a, emb_b, rel1, We,
                                                eat1, a_attn, row1, col1, sc1, m_b);
  edge_score_kernel<<<EGRID, EBLK, 0, stream>>>(q_a, k_b, emb_b, emb_a, rel2, We,
                                                eat2, a_attn, row2, col2, sc2, m_a);

  edge_agg_kernel<<<EGRID, EBLK, 0, stream>>>(sc1, m_b, v_a, rel1, We, eat1,
                                              row1, col1, s_b, agg_b);
  edge_agg_kernel<<<EGRID, EBLK, 0, stream>>>(sc2, m_a, v_b, rel2, We, eat2,
                                              row2, col2, s_a, agg_a);

  // outputs: [fin_a ; fin_b] -> [2N, 32] (WMMA)
  out_gemm_kernel<<<TILES, 32, 0, stream>>>(agg_a, s_a, x_a, Wo_a, bo_a, Wr_a, out);
  out_gemm_kernel<<<TILES, 32, 0, stream>>>(agg_b, s_b, x_b, Wo_b, bo_b, Wr_b,
                                            out + (size_t)N_NODES * OC);
}
